// RisingTideAttentionV2_18339510354290
// MI455X (gfx1250) — compile-verified, matrix operand order checked by reading, not run on hardware
//
#include <hip/hip_runtime.h>
#include <hip/hip_bf16.h>

typedef __attribute__((ext_vector_type(16))) __bf16 v16bf;
typedef __attribute__((ext_vector_type(8)))  __bf16 v8bf;
typedef __attribute__((ext_vector_type(8)))  float  v8f;

#define DDIM 128
// NEURON_SCALE = 0.05 + (0.8-0.05) * (1/600)
#define NSCALE 0.05125f

__device__ __forceinline__ unsigned short f32_to_bf16(float f) {
  unsigned int u = __float_as_uint(f);
  u += 0x7FFFu + ((u >> 16) & 1u);   // round to nearest even
  return (unsigned short)(u >> 16);
}

union BF16x16 { v16bf v; v8bf h[2]; };

// Load one 16-bit WMMA A/B operand (16x32 / 32x16 tile) for this lane.
// rowbase points at this lane's row, stored row-major with K contiguous.
// Layout per ISA 7.12.2:
//   lanes 0-15 (g=0): VGPR0-3 hold K=colbase+0..7,  VGPR4-7 hold K=colbase+16..23
//   lanes 16-31(g=1): VGPR0-3 hold K=colbase+8..15, VGPR4-7 hold K=colbase+24..31
__device__ __forceinline__ v16bf load_op_bf16(const unsigned short* rowbase,
                                              int colbase, int g) {
  const int o = colbase + 8 * g;
  BF16x16 r;
  r.h[0] = *reinterpret_cast<const v8bf*>(rowbase + o);
  r.h[1] = *reinterpret_cast<const v8bf*>(rowbase + o + 16);
  return r.v;
}

// Generic shared pointer -> LDS byte offset (shared aperture keeps the LDS
// offset in the low 32 bits; see ISA 10.2).
__device__ __forceinline__ unsigned lds_off_u32(const void* p) {
  return (unsigned)(size_t)p;
}

// gfx1250 async copy global -> LDS, tracked by ASYNCcnt.
__device__ __forceinline__ void async_ld_b128(unsigned lds_byte_off, const void* gsrc) {
  asm volatile("global_load_async_to_lds_b128 %0, %1, off"
               :: "v"(lds_byte_off), "v"(gsrc)
               : "memory");
}

__device__ __forceinline__ void wait_asynccnt0() {
  asm volatile("s_wait_asynccnt 0x0" ::: "memory");
}

// ---------- prepass: x -> bf16, x2 row norms (one wave per row of 128) ----------
__global__ __launch_bounds__(256) void prep_x(const float* __restrict__ x,
                                              unsigned short* __restrict__ xb,
                                              float* __restrict__ x2, int M) {
  int wid  = (int)((blockIdx.x * blockDim.x + threadIdx.x) >> 5);
  int lane = threadIdx.x & 31;
  if (wid >= M) return;
  const float4 v = reinterpret_cast<const float4*>(x + (size_t)wid * DDIM)[lane];
  float ss = v.x * v.x + v.y * v.y + v.z * v.z + v.w * v.w;
  ushort4 b;
  b.x = f32_to_bf16(v.x); b.y = f32_to_bf16(v.y);
  b.z = f32_to_bf16(v.z); b.w = f32_to_bf16(v.w);
  reinterpret_cast<ushort4*>(xb + (size_t)wid * DDIM)[lane] = b;
  ss += __shfl_xor(ss, 16, 32);
  ss += __shfl_xor(ss, 8, 32);
  ss += __shfl_xor(ss, 4, 32);
  ss += __shfl_xor(ss, 2, 32);
  ss += __shfl_xor(ss, 1, 32);
  if (lane == 0) x2[wid] = ss;
}

// ---- prepass: positions -> bf16 + p2, values -> bf16 transposed [D][N], inv temp ----
__global__ __launch_bounds__(256) void prep_n(const float* __restrict__ pos,
                                              const float* __restrict__ val,
                                              const float* __restrict__ tmp,
                                              unsigned short* __restrict__ pb,
                                              unsigned short* __restrict__ vT,
                                              float* __restrict__ p2,
                                              float* __restrict__ ie, int N) {
  int wid  = (int)((blockIdx.x * blockDim.x + threadIdx.x) >> 5);
  int lane = threadIdx.x & 31;
  if (wid >= N) return;
  const float4 pv = reinterpret_cast<const float4*>(pos + (size_t)wid * DDIM)[lane];
  float ss = pv.x * pv.x + pv.y * pv.y + pv.z * pv.z + pv.w * pv.w;
  ushort4 b;
  b.x = f32_to_bf16(pv.x); b.y = f32_to_bf16(pv.y);
  b.z = f32_to_bf16(pv.z); b.w = f32_to_bf16(pv.w);
  reinterpret_cast<ushort4*>(pb + (size_t)wid * DDIM)[lane] = b;
  ss += __shfl_xor(ss, 16, 32);
  ss += __shfl_xor(ss, 8, 32);
  ss += __shfl_xor(ss, 4, 32);
  ss += __shfl_xor(ss, 2, 32);
  ss += __shfl_xor(ss, 1, 32);
  if (lane == 0) {
    p2[wid] = ss;
    ie[wid] = 1.0f / ((fabsf(tmp[wid]) + 0.1f) * NSCALE);
  }
  const float4 vv = reinterpret_cast<const float4*>(val + (size_t)wid * DDIM)[lane];
  int d = lane * 4;
  vT[(size_t)(d + 0) * N + wid] = f32_to_bf16(vv.x);
  vT[(size_t)(d + 1) * N + wid] = f32_to_bf16(vv.y);
  vT[(size_t)(d + 2) * N + wid] = f32_to_bf16(vv.z);
  vT[(size_t)(d + 3) * N + wid] = f32_to_bf16(vv.w);
}

// ---------------- fused main kernel: one wave owns 16 output rows ----------------
__global__ __launch_bounds__(128) void tide_main(
    const unsigned short* __restrict__ xb,   // [M][128] bf16
    const unsigned short* __restrict__ pb,   // [N][128] bf16
    const unsigned short* __restrict__ vT,   // [128][N] bf16 (values transposed)
    const float* __restrict__ x2,            // [M]
    const float* __restrict__ p2,            // [N]
    const float* __restrict__ ie,            // [N] 1/effective_temp
    float* __restrict__ out, int N) {
  const int tid  = threadIdx.x;
  const int lane = tid & 31;
  const int wave = tid >> 5;
  const int l16  = lane & 15;
  const int g    = lane >> 4;
  const int mbase = ((int)blockIdx.x * 4 + wave) * 16;

  // Double-buffered async stages shared by the whole block (36 KB total LDS)
  __shared__ unsigned short pstage[2][32 * DDIM];  // positions chunk [32 n][128 k]
  __shared__ unsigned short vstage[2][DDIM * 32];  // valuesT chunk   [128 d][32 n]
  __shared__ unsigned short wlds[4][16 * 32];      // per-wave W tile
  unsigned short* wl = &wlds[wave][0];

  // Issue one chunk's async global->LDS copies (8 KB + 8 KB, 8 b128 per thread)
  auto issue_stage = [&](int b, int nb) {
    const unsigned short* psrc = pb + (size_t)nb * DDIM;      // 32 rows, contiguous 8 KB
    unsigned pdst = lds_off_u32(&pstage[b][0]);
#pragma unroll
    for (int i = 0; i < 4; i++)
      async_ld_b128(pdst + (unsigned)(tid + i * 128) * 16u,
                    psrc + (size_t)(tid + i * 128) * 8);
    const unsigned short* vsrc = vT + (size_t)tid * N + nb;   // row d=tid, 64 B
    unsigned vdst = lds_off_u32(&vstage[b][tid * 32]);
#pragma unroll
    for (int i = 0; i < 4; i++)
      async_ld_b128(vdst + (unsigned)i * 16u, vsrc + i * 8);
  };

  // Resident A tiles: 16 rows x 128 K of x, split into 4 K-steps of 32
  v16bf Ax[4];
#pragma unroll
  for (int kk = 0; kk < 4; kk++)
    Ax[kk] = load_op_bf16(xb + (size_t)(mbase + l16) * DDIM, kk * 32, g);

  float x2v[8];
#pragma unroll
  for (int r = 0; r < 8; r++) x2v[r] = x2[mbase + r + 8 * g];

  const v8f vzero = {};
  v8f O[8];
#pragma unroll
  for (int t = 0; t < 8; t++) O[t] = vzero;
  float rs[8] = {0.f, 0.f, 0.f, 0.f, 0.f, 0.f, 0.f, 0.f};

  int buf = 0;
  issue_stage(0, 0);

  for (int nb = 0; nb < N; nb += 32) {
    wait_asynccnt0();     // this wave's async copies of stage[buf] done
    __syncthreads();      // all waves' copies done; all readers of stage[buf^1] done
    if (nb + 32 < N) issue_stage(buf ^ 1, nb + 32);   // overlap next copy w/ compute

    const unsigned short* ps = &pstage[buf][0];
    const unsigned short* vs = &vstage[buf][0];

    // ---- GEMM1: S[16x32] = x_tile @ positions[nb:nb+32]^T (B operands from LDS) ----
    v8f S0 = vzero, S1 = vzero;
#pragma unroll
    for (int kk = 0; kk < 4; kk++) {
      v16bf B0 = load_op_bf16(ps + l16 * DDIM,        kk * 32, g);
      v16bf B1 = load_op_bf16(ps + (16 + l16) * DDIM, kk * 32, g);
      S0 = __builtin_amdgcn_wmma_f32_16x16x32_bf16(false, Ax[kk], false, B0,
                                                   (short)0, S0, false, false);
      S1 = __builtin_amdgcn_wmma_f32_16x16x32_bf16(false, Ax[kk], false, B1,
                                                   (short)0, S1, false, false);
    }

    // ---- fused elementwise: d2 -> dist -> exp, rowsum, pack W to LDS ----
    const float p2v0 = p2[nb + l16],      p2v1 = p2[nb + 16 + l16];
    const float ie0  = ie[nb + l16],      ie1  = ie[nb + 16 + l16];
#pragma unroll
    for (int r = 0; r < 8; r++) {
      float d0 = fmaxf(x2v[r] + p2v0 - 2.0f * S0[r], 0.0f);
      float w0 = __expf(-sqrtf(d0) * ie0);
      float d1 = fmaxf(x2v[r] + p2v1 - 2.0f * S1[r], 0.0f);
      float w1 = __expf(-sqrtf(d1) * ie1);
      rs[r] += w0 + w1;
      wl[(r + 8 * g) * 32 + l16]      = f32_to_bf16(w0);   // C-layout -> row-major LDS
      wl[(r + 8 * g) * 32 + 16 + l16] = f32_to_bf16(w1);
    }
    // LDS round-trip converts C-layout to A-layout; wave-private tile, but make
    // the store->cross-lane-load ordering explicit.
    asm volatile("s_wait_dscnt 0x0" ::: "memory");
    v16bf Wa = load_op_bf16(wl + l16 * 32, 0, g);

    // ---- GEMM2: O[16x128] += W[16x32] @ values[nb:nb+32][0:128] (B from LDS) ----
#pragma unroll
    for (int t = 0; t < 8; t++) {
      v16bf Bv = load_op_bf16(vs + (t * 16 + l16) * 32, 0, g);
      O[t] = __builtin_amdgcn_wmma_f32_16x16x32_bf16(false, Wa, false, Bv,
                                                     (short)0, O[t], false, false);
    }
    buf ^= 1;
  }

  // ---- rowsum reduce across each 16-lane half (matches C-layout halves) ----
#pragma unroll
  for (int r = 0; r < 8; r++) {
    float s = rs[r];
    s += __shfl_xor(s, 1, 32);
    s += __shfl_xor(s, 2, 32);
    s += __shfl_xor(s, 4, 32);
    s += __shfl_xor(s, 8, 32);
    rs[r] = 1.0f / (s + 1e-8f);
  }

  // ---- normalized store: element (t, r, lane) -> out[mbase + r + 8g][t*16 + l16] ----
#pragma unroll
  for (int t = 0; t < 8; t++)
#pragma unroll
    for (int r = 0; r < 8; r++)
      out[(size_t)(mbase + r + 8 * g) * DDIM + t * 16 + l16] = O[t][r] * rs[r];
}

extern "C" void kernel_launch(void* const* d_in, const int* in_sizes, int n_in,
                              void* d_out, int out_size, void* d_ws, size_t ws_size,
                              hipStream_t stream) {
  const float* x   = (const float*)d_in[0];
  const float* pos = (const float*)d_in[1];
  const float* val = (const float*)d_in[2];
  const float* tmp = (const float*)d_in[3];
  float* out = (float*)d_out;

  const int D = DDIM;
  const int M = in_sizes[0] / D;   // 16384 for reference shapes
  const int N = in_sizes[3];       // 4096

  // Carve workspace (needs ~6.2 MB for reference shapes)
  char* w = (char*)d_ws;
  size_t off = 0;
  auto carve = [&](size_t bytes) -> char* {
    char* p = w + off;
    off += (bytes + 255) & ~(size_t)255;
    return p;
  };
  unsigned short* xb = (unsigned short*)carve((size_t)M * D * 2);
  unsigned short* pb = (unsigned short*)carve((size_t)N * D * 2);
  unsigned short* vT = (unsigned short*)carve((size_t)N * D * 2);
  float* x2 = (float*)carve((size_t)M * 4);
  float* p2 = (float*)carve((size_t)N * 4);
  float* ie = (float*)carve((size_t)N * 4);
  (void)ws_size; (void)n_in; (void)out_size;

  prep_x<<<(M + 7) / 8, 256, 0, stream>>>(x, xb, x2, M);
  prep_n<<<(N + 7) / 8, 256, 0, stream>>>(pos, val, tmp, pb, vT, p2, ie, N);
  // 4 waves/block, 16 rows/wave -> 64 rows/block
  tide_main<<<M / 64, 128, 0, stream>>>(xb, pb, vT, x2, p2, ie, out, N);
}
